// Attention_2662879723657
// MI455X (gfx1250) — matrix operand order repaired and checked by reading, not compiled, not run on hardware
//
#include <hip/hip_runtime.h>
#include <hip/hip_bf16.h>
#include <math.h>

typedef __attribute__((ext_vector_type(16))) __bf16 v16bf;
typedef __attribute__((ext_vector_type(8)))  float  v8f;

#define BATCH 512
#define NPIX  196
#define EDIM  512
#define ADIM  512
#define HDIM  512
#define NPIX_PAD 208      // 13 * 16
#define MTILES 13
#define CTX_ELEMS (BATCH * EDIM)

// LDS A-tile: 16 rows x 512 bf16, padded to 520 halfwords (260 dwords) per row.
// 260 % 64 == 4 -> the 16 lanes of a half-wave start on distinct 4-dword bank
// groups during ds_load_b128 fragment reads (conflict-free).
#define AT_STRIDE_H 520
#define AT_STRIDE_U 260

union ABFrag { uint4 q[2]; unsigned u[8]; v16bf v; };

static __device__ __forceinline__ unsigned pack_bf16_trunc(float lo, float hi) {
    unsigned a = __float_as_uint(lo);
    unsigned b = __float_as_uint(hi);
    return (b & 0xFFFF0000u) | (a >> 16);
}
static __device__ __forceinline__ unsigned short bf16_rne(float f) {
    unsigned u = __float_as_uint(f);
    unsigned r = u + 0x7FFFu + ((u >> 16) & 1u);
    return (unsigned short)(r >> 16);
}

// ---------------------------------------------------------------------------
// Kernel 0: W_enc [E][A] f32  ->  Wt [A][E] bf16 (transposed, round-to-nearest)
// ---------------------------------------------------------------------------
__global__ void wenc_to_bf16_t(const float* __restrict__ W_enc,
                               unsigned short* __restrict__ Wt) {
    const int a  = blockIdx.x;        // output row (attention dim)
    const int e0 = threadIdx.x * 2;   // two columns per thread
    float f0 = W_enc[(size_t)e0 * ADIM + a];
    float f1 = W_enc[(size_t)(e0 + 1) * ADIM + a];
    unsigned p = ((unsigned)bf16_rne(f1) << 16) | (unsigned)bf16_rne(f0);
    *(unsigned*)(Wt + (size_t)a * EDIM + e0) = p;
}

// ---------------------------------------------------------------------------
// Kernel 1: S[b][a] = decoder_hidden[b] @ W_dec[:,a] + b_dec[a] + b_enc[a]
// ---------------------------------------------------------------------------
__global__ void dec_bias_kernel(const float* __restrict__ hid,
                                const float* __restrict__ W_dec,
                                const float* __restrict__ b_dec,
                                const float* __restrict__ b_enc,
                                float* __restrict__ S) {
    __shared__ float h[HDIM];
    const int b = blockIdx.x, tid = threadIdx.x;
    h[tid]       = hid[(size_t)b * HDIM + tid];
    h[tid + 256] = hid[(size_t)b * HDIM + tid + 256];
    __syncthreads();
    #pragma unroll
    for (int j = 0; j < 2; ++j) {
        const int a = tid + j * 256;
        float acc = 0.f;
        #pragma unroll 8
        for (int k = 0; k < HDIM; ++k)
            acc = fmaf(h[k], W_dec[(size_t)k * ADIM + a], acc);
        S[(size_t)b * ADIM + a] = acc + b_dec[a] + b_enc[a];
    }
}

// ---------------------------------------------------------------------------
// Kernel 2: fused  att_enc GEMM (bf16 WMMA) + relu-dot + softmax + context
// One workgroup (8 waves) per batch row b.
// ---------------------------------------------------------------------------
__global__ void __launch_bounds__(256) attn_main(
    const float* __restrict__ enc, const unsigned short* __restrict__ Wt,
    const float* __restrict__ S, const float* __restrict__ w_full,
    const float* __restrict__ b_full, float* __restrict__ out) {

    __shared__ unsigned Atile_u[16 * AT_STRIDE_U];   // 16.25 KB bf16 A tile
    __shared__ float energy[NPIX_PAD];
    __shared__ float alpha [NPIX_PAD];
    __shared__ float red[8];

    const int b    = blockIdx.x;
    const int tid  = threadIdx.x;
    const int wave = tid >> 5;
    const int lane = tid & 31;
    const int half = lane >> 4;
    const int lq   = lane & 15;

    if (tid < NPIX_PAD) energy[tid] = b_full[0];

    const float* encB = enc + (size_t)b * NPIX * EDIM;
    const float* Sb   = S + (size_t)b * ADIM;
    const int a0 = wave * 64;   // this wave covers attention cols [a0, a0+64)

    // Loop-invariant epilogue vectors: S[a] + both biases, and w_full[a]
    float sv[4], wv[4];
    #pragma unroll
    for (int t = 0; t < 4; ++t) {
        const int ac = a0 + t * 16 + lq;
        sv[t] = Sb[ac];
        wv[t] = w_full[ac];
    }

    // B row base pointers (bf16, transposed W_enc) for this lane's column
    const unsigned short* B0 = Wt + (size_t)(a0 +  0 + lq) * EDIM + half * 16;
    const unsigned short* B1 = Wt + (size_t)(a0 + 16 + lq) * EDIM + half * 16;
    const unsigned short* B2 = Wt + (size_t)(a0 + 32 + lq) * EDIM + half * 16;
    const unsigned short* B3 = Wt + (size_t)(a0 + 48 + lq) * EDIM + half * 16;

    __syncthreads();   // energy[] initialized

    // ---------------- Phase 1: energy[p] via bf16 WMMA ----------------
    for (int mt = 0; mt < MTILES; ++mt) {
        // --- cooperative stage: convert A tile (16 x 512 f32 -> bf16) into LDS
        #pragma unroll
        for (int i = 0; i < 16; ++i) {
            const int idx  = tid + i * 256;    // 0..4095 packed-uint index
            const int row  = idx >> 8;         // 256 uints (512 bf16) per row
            const int col2 = idx & 255;
            int prow = mt * 16 + row;
            prow = prow > (NPIX - 1) ? (NPIX - 1) : prow;   // clamp padded rows
            const float2 f = *(const float2*)(encB + (size_t)prow * EDIM + col2 * 2);
            Atile_u[row * AT_STRIDE_U + col2] = pack_bf16_trunc(f.x, f.y);
        }
        __syncthreads();

        const unsigned short* Ar =
            (const unsigned short*)Atile_u + (size_t)lq * AT_STRIDE_H + half * 16;

        v8f c0 = {}, c1 = {}, c2 = {}, c3 = {};

        #pragma unroll 4
        for (int kk = 0; kk < 16; ++kk) {
            // Issue ALL four B fragment loads into distinct registers first so
            // the scheduler can use partial s_wait_loadcnt values and overlap
            // later B loads with earlier WMMAs (loads complete in order).
            ABFrag b0f, b1f, b2f, b3f, a;
            b0f.q[0] = *(const uint4*)(B0 + kk * 32);
            b0f.q[1] = *(const uint4*)(B0 + kk * 32 + 8);
            b1f.q[0] = *(const uint4*)(B1 + kk * 32);
            b1f.q[1] = *(const uint4*)(B1 + kk * 32 + 8);
            b2f.q[0] = *(const uint4*)(B2 + kk * 32);
            b2f.q[1] = *(const uint4*)(B2 + kk * 32 + 8);
            b3f.q[0] = *(const uint4*)(B3 + kk * 32);
            b3f.q[1] = *(const uint4*)(B3 + kk * 32 + 8);

            // A fragment from LDS: 16 contiguous K bf16 per half (k = 16h + s, matches B)
            a.q[0] = *(const uint4*)(Ar + kk * 32);
            a.q[1] = *(const uint4*)(Ar + kk * 32 + 8);

            c0 = __builtin_amdgcn_wmma_f32_16x16x32_bf16(false, a.v, false, b0f.v, (short)0, c0, false, false);
            c1 = __builtin_amdgcn_wmma_f32_16x16x32_bf16(false, a.v, false, b1f.v, (short)0, c1, false, false);
            c2 = __builtin_amdgcn_wmma_f32_16x16x32_bf16(false, a.v, false, b2f.v, (short)0, c2, false, false);
            c3 = __builtin_amdgcn_wmma_f32_16x16x32_bf16(false, a.v, false, b3f.v, (short)0, c3, false, false);
        }

        // Epilogue: relu(C + S[a]) * w_full[a], accumulated per (r, half) = p
        float ep[8];
        #pragma unroll
        for (int r = 0; r < 8; ++r) ep[r] = 0.f;
        #pragma unroll
        for (int r = 0; r < 8; ++r) { float v = c0[r] + sv[0]; v = v > 0.f ? v : 0.f; ep[r] = fmaf(v, wv[0], ep[r]); }
        #pragma unroll
        for (int r = 0; r < 8; ++r) { float v = c1[r] + sv[1]; v = v > 0.f ? v : 0.f; ep[r] = fmaf(v, wv[1], ep[r]); }
        #pragma unroll
        for (int r = 0; r < 8; ++r) { float v = c2[r] + sv[2]; v = v > 0.f ? v : 0.f; ep[r] = fmaf(v, wv[2], ep[r]); }
        #pragma unroll
        for (int r = 0; r < 8; ++r) { float v = c3[r] + sv[3]; v = v > 0.f ? v : 0.f; ep[r] = fmaf(v, wv[3], ep[r]); }

        // Reduce across the 16 N-lanes of each half, then ds_add into energy[p]
        #pragma unroll
        for (int r = 0; r < 8; ++r) {
            float v = ep[r];
            v += __shfl_xor(v, 1);
            v += __shfl_xor(v, 2);
            v += __shfl_xor(v, 4);
            v += __shfl_xor(v, 8);
            if (lq == 0) {
                int p = mt * 16 + half * 8 + r;
                if (p < NPIX) atomicAdd(&energy[p], v);
            }
        }
        __syncthreads();   // A tile consumed + energies posted before restage
    }

    // ---------------- Phase 2: softmax over p ----------------
    float ev = (tid < NPIX) ? energy[tid] : -3.0e38f;
    float m = ev;
    #pragma unroll
    for (int mask = 16; mask >= 1; mask >>= 1) m = fmaxf(m, __shfl_xor(m, mask));
    if (lane == 0) red[wave] = m;
    __syncthreads();
    if (tid == 0) {
        float mm = red[0];
        #pragma unroll
        for (int i = 1; i < 8; ++i) mm = fmaxf(mm, red[i]);
        red[0] = mm;
    }
    __syncthreads();
    const float mx = red[0];
    float ex = (tid < NPIX) ? __expf(ev - mx) : 0.f;
    __syncthreads();
    float sum = ex;
    #pragma unroll
    for (int mask = 16; mask >= 1; mask >>= 1) sum += __shfl_xor(sum, mask);
    if (lane == 0) red[wave] = sum;
    __syncthreads();
    if (tid == 0) {
        float ss = 0.f;
        #pragma unroll
        for (int i = 0; i < 8; ++i) ss += red[i];
        red[0] = ss;
    }
    __syncthreads();
    const float inv = 1.0f / red[0];
    const float al = ex * inv;
    if (tid < NPIX) {
        alpha[tid] = al;
        out[CTX_ELEMS + (size_t)b * NPIX + tid] = al;
    }
    __syncthreads();

    // ---------------- Phase 3: context[b][e] = sum_p alpha[p] * enc[b][p][e]
    float acc0 = 0.f, acc1 = 0.f;
    #pragma unroll 4
    for (int p = 0; p < NPIX; ++p) {
        const float a = alpha[p];
        const float* row = encB + (size_t)p * EDIM;
        acc0 = fmaf(a, row[tid], acc0);
        acc1 = fmaf(a, row[tid + 256], acc1);
    }
    out[(size_t)b * EDIM + tid]       = acc0;
    out[(size_t)b * EDIM + tid + 256] = acc1;
}

// ---------------------------------------------------------------------------
extern "C" void kernel_launch(void* const* d_in, const int* in_sizes, int n_in,
                              void* d_out, int out_size, void* d_ws, size_t ws_size,
                              hipStream_t stream) {
    const float* enc   = (const float*)d_in[0];
    const float* hid   = (const float*)d_in[1];
    const float* W_enc = (const float*)d_in[2];
    const float* b_enc = (const float*)d_in[3];
    const float* W_dec = (const float*)d_in[4];
    const float* b_dec = (const float*)d_in[5];
    const float* wfull = (const float*)d_in[6];
    const float* bfull = (const float*)d_in[7];
    float* out = (float*)d_out;

    unsigned short* Wt = (unsigned short*)d_ws;                          // 512 KB
    float* S = (float*)((char*)d_ws + (size_t)ADIM * EDIM * sizeof(unsigned short)); // 1 MB

    wenc_to_bf16_t<<<dim3(ADIM),  dim3(256), 0, stream>>>(W_enc, Wt);
    dec_bias_kernel<<<dim3(BATCH), dim3(256), 0, stream>>>(hid, W_dec, b_dec, b_enc, S);
    attn_main<<<dim3(BATCH), dim3(256), 0, stream>>>(enc, Wt, S, wfull, bfull, out);
}
